// Block_19121194402322
// MI455X (gfx1250) — matrix operand order repaired and checked by reading, not compiled, not run on hardware
//
#include <hip/hip_runtime.h>
#include <hip/hip_bf16.h>
#include <math.h>

// ---------------------------------------------------------------------------
// Transformer block on MI455X (gfx1250), exploiting the reference's algebra:
//   * attn einsum sums over k AND v  ->  out[b,n,q,d] = sum_s v[b,n,s,d]
//     (softmax rows sum to 1) -> Q/K GEMMs + attention are dead code.
//   * out2[b,s,:] is s-invariant -> LN1 is an 8x768 per-batch vector.
// Remaining work: 3 GEMMs (~48 GFLOP) in f16 WMMA w/ f32 accumulate, fed by
// CDNA5 async global->LDS (ASYNCcnt) with double-buffered LDS tiles.
// Workspace requirement: ~82 MB.
// ---------------------------------------------------------------------------

typedef __attribute__((ext_vector_type(16))) _Float16 v16h;
typedef __attribute__((ext_vector_type(8)))  _Float16 v8h;
typedef __attribute__((ext_vector_type(8)))  float    v8f;

#define EMBED 768
#define HID   1536
#define BS    8
#define SEQ   1024
#define ROWS  (BS * SEQ)          // 8192
#define LN_EPS 1e-5f

__device__ __forceinline__ v16h cat16(v8h lo, v8h hi) {
    return __builtin_shufflevector(lo, hi, 0,1,2,3,4,5,6,7,8,9,10,11,12,13,14,15);
}
__device__ __forceinline__ float gelu_exact(float v) {
    return 0.5f * v * (1.0f + erff(v * 0.70710678118654752f));
}

// ---- CDNA5 async global->LDS (tracked by ASYNCcnt, writes LDS directly) ----
// vdst = per-lane LDS byte address (low 32 bits of the generic pointer: the
// AS3->flat addrspacecast puts the LDS offset in addr[31:0]; HW adds LDS_BASE).
__device__ __forceinline__ void async_ld_b128(void* lds_dst, const void* gsrc) {
    uint32_t l = (uint32_t)(uintptr_t)lds_dst;
    asm volatile("global_load_async_to_lds_b128 %0, %1, off"
                 :: "v"(l), "v"(gsrc)
                 : "memory");
}
#define S_WAIT_ASYNCCNT(n) asm volatile("s_wait_asynccnt %0" :: "n"(n) : "memory")

// ------------------------- conversion kernels ------------------------------
__global__ __launch_bounds__(256) void cvt_f16_kernel(const float* __restrict__ s,
                                                      _Float16* __restrict__ d, int n) {
    int i = blockIdx.x * 256 + threadIdx.x;
    if (i < n) d[i] = (_Float16)s[i];
}

// W (K x N, row-major) -> Wt (N x K, row-major), f32 -> f16
__global__ __launch_bounds__(256) void cvt_tr_kernel(const float* __restrict__ W,
                                                     _Float16* __restrict__ Wt,
                                                     int K, int N) {
    int o = blockIdx.x * 256 + threadIdx.x;          // output index: n*K + k
    if (o < K * N) {
        int n = o / K, k = o - n * K;
        Wt[o] = (_Float16)W[(size_t)k * N + n];
    }
}

// --------------------------- WMMA GEMM -------------------------------------
// C[M,N] = A[M,K] (f16 row-major) * Bt[N,K]^T (f16, N-major = pre-transposed B)
// EPI: 0 = store f32; 1 = bias+GELU, store f16; 2 = bias+GELU, store f32
#define TBM 128
#define TBN 128
#define TBK 32
#define LDP 40   // LDS row stride in halfs (32 + 8 pad -> conflict-free frags)

template<int EPI>
__global__ __launch_bounds__(256)
void gemm_wmma(const _Float16* __restrict__ A, const _Float16* __restrict__ Bt,
               const float* __restrict__ bias, void* __restrict__ Cout,
               int M, int N, int K) {
    __shared__ _Float16 As[2][TBM][LDP];   // double-buffered: 2 x 10 KB
    __shared__ _Float16 Bs[2][TBN][LDP];   // double-buffered: 2 x 10 KB

    const int tid  = threadIdx.x;
    const int lane = tid & 31;
    const int wid  = tid >> 5;            // 8 waves
    const int wm0  = (wid >> 2) * 64;     // wave M offset in block tile
    const int wn0  = (wid & 3) * 32;      // wave N offset in block tile
    const int bm0  = blockIdx.y * TBM;
    const int bn0  = blockIdx.x * TBN;
    const int hs   = lane >> 4;           // half-select (0: lanes 0-15, 1: 16-31)
    const int l16  = lane & 15;

    v8f acc[4][2];
#pragma unroll
    for (int mt = 0; mt < 4; ++mt)
#pragma unroll
        for (int nt = 0; nt < 2; ++nt) acc[mt][nt] = (v8f)(0.0f);

    // global -> LDS staging: each thread moves 32B of A tile and 32B of B tile
    const int ldr = tid >> 1;             // 0..127 (tile row)
    const int ldc = (tid & 1) * 16;       // 0 or 16 (halfs)
    const _Float16* gA = A  + (size_t)(bm0 + ldr) * K + ldc;
    const _Float16* gB = Bt + (size_t)(bn0 + ldr) * K + ldc;

    const int iters = K / TBK;

    // prime the pipe: tile 0 -> buffer 0 (4 async ops per thread per tile)
    async_ld_b128(&As[0][ldr][ldc],     gA);
    async_ld_b128(&As[0][ldr][ldc + 8], gA + 8);
    async_ld_b128(&Bs[0][ldr][ldc],     gB);
    async_ld_b128(&Bs[0][ldr][ldc + 8], gB + 8);

    for (int kt = 0; kt < iters; ++kt) {
        const int buf = kt & 1;
        // prefetch next tile into the other buffer (dummy re-load of tile 0 on
        // the last iteration keeps the wait immediate constant; buf^1 is dead
        // then, and the trailing barrier of kt-1 made buf^1 safe to overwrite)
        const int kn = (kt + 1 < iters) ? (kt + 1) * TBK : 0;
        async_ld_b128(&As[buf ^ 1][ldr][ldc],     gA + kn);
        async_ld_b128(&As[buf ^ 1][ldr][ldc + 8], gA + kn + 8);
        async_ld_b128(&Bs[buf ^ 1][ldr][ldc],     gB + kn);
        async_ld_b128(&Bs[buf ^ 1][ldr][ldc + 8], gB + kn + 8);

        // async completes in order: <=4 outstanding means tile kt has landed
        S_WAIT_ASYNCCNT(4);
        __syncthreads();

        // B fragments: lane holds column n=l16, K-chunk hs*16..hs*16+15
        v16h bfr[2];
#pragma unroll
        for (int nt = 0; nt < 2; ++nt) {
            v8h lo = *(const v8h*)&Bs[buf][wn0 + nt * 16 + l16][hs * 16];
            v8h hi = *(const v8h*)&Bs[buf][wn0 + nt * 16 + l16][hs * 16 + 8];
            bfr[nt] = cat16(lo, hi);
        }
        // A fragments: lane holds row m=l16, K-chunk hs*16..hs*16+15
#pragma unroll
        for (int mt = 0; mt < 4; ++mt) {
            v8h lo = *(const v8h*)&As[buf][wm0 + mt * 16 + l16][hs * 16];
            v8h hi = *(const v8h*)&As[buf][wm0 + mt * 16 + l16][hs * 16 + 8];
            v16h afr = cat16(lo, hi);
            acc[mt][0] = __builtin_amdgcn_wmma_f32_16x16x32_f16(
                false, afr, false, bfr[0], (short)0, acc[mt][0], false, false);
            acc[mt][1] = __builtin_amdgcn_wmma_f32_16x16x32_f16(
                false, afr, false, bfr[1], (short)0, acc[mt][1], false, false);
        }
        __syncthreads();   // all waves done with buf before it is overwritten
    }

    // epilogue: lane holds column n, rows m = base + hs*8 + j  (j = vgpr index)
#pragma unroll
    for (int mt = 0; mt < 4; ++mt) {
#pragma unroll
        for (int nt = 0; nt < 2; ++nt) {
            const int n  = bn0 + wn0 + nt * 16 + l16;
            const int mb = bm0 + wm0 + mt * 16 + hs * 8;
            float bv = 0.0f;
            if constexpr (EPI != 0) bv = bias[n];
#pragma unroll
            for (int j = 0; j < 8; ++j) {
                float v = acc[mt][nt][j] + bv;
                if constexpr (EPI != 0) v = gelu_exact(v);
                if constexpr (EPI == 1)
                    ((_Float16*)Cout)[(size_t)(mb + j) * N + n] = (_Float16)v;
                else
                    ((float*)Cout)[(size_t)(mb + j) * N + n] = v;
            }
        }
    }
}

// ---------------- vsum: out[b, n*64+d] = sum_j V[b, n*65536 + j*64 + d] -----
__global__ __launch_bounds__(256) void vsum_kernel(const float* __restrict__ V,
                                                   float* __restrict__ vs) {
    const int n = blockIdx.x, b = blockIdx.y;
    const int t = threadIdx.x;
    const int d = t & 63, g = t >> 6;    // 4 groups of 64
    const float* Vb = V + (size_t)b * (SEQ * EMBED) + (size_t)n * 65536;
    float s = 0.0f;
    for (int j = g; j < 1024; j += 4) s += Vb[j * 64 + d];
    __shared__ float red[256];
    red[t] = s;
    __syncthreads();
    if (g == 0)
        vs[b * EMBED + n * 64 + d] = red[d] + red[64 + d] + red[128 + d] + red[192 + d];
}

// ---------------- LN over one 768-vector per batch (broadcast LN1) ----------
__global__ __launch_bounds__(256) void ln_vec_kernel(const float* __restrict__ vs,
                                                     const float* __restrict__ g,
                                                     const float* __restrict__ be,
                                                     float* __restrict__ out) {
    const int b = blockIdx.x, t = threadIdx.x;
    const float* r = vs + b * EMBED;
    __shared__ float red[256];
    float s = 0.0f;
    for (int e = t; e < EMBED; e += 256) s += r[e];
    red[t] = s; __syncthreads();
    for (int st = 128; st > 0; st >>= 1) { if (t < st) red[t] += red[t + st]; __syncthreads(); }
    const float mu = red[0] * (1.0f / EMBED);
    __syncthreads();
    float s2 = 0.0f;
    for (int e = t; e < EMBED; e += 256) { float d = r[e] - mu; s2 += d * d; }
    red[t] = s2; __syncthreads();
    for (int st = 128; st > 0; st >>= 1) { if (t < st) red[t] += red[t + st]; __syncthreads(); }
    const float rinv = rsqrtf(red[0] * (1.0f / EMBED) + LN_EPS);
    for (int e = t; e < EMBED; e += 256)
        out[b * EMBED + e] = (r[e] - mu) * rinv * g[e] + be[e];
}

// -------- a = x + broadcast(ln1), converted to f16 for the next GEMM --------
__global__ __launch_bounds__(256) void addbc_f16_kernel(const float* __restrict__ x,
                                                        const float* __restrict__ lnb,
                                                        _Float16* __restrict__ a16) {
    int i = blockIdx.x * 256 + threadIdx.x;     // < 8*1024*768
    int e = i % EMBED;
    int b = i / (SEQ * EMBED);
    a16[i] = (_Float16)(x[i] + lnb[b * EMBED + e]);
}

// --------------- out = x + LN(m) per row (8192 rows of 768) -----------------
__global__ __launch_bounds__(256) void final_ln_kernel(const float* __restrict__ x,
                                                       const float* __restrict__ m,
                                                       const float* __restrict__ g,
                                                       const float* __restrict__ be,
                                                       float* __restrict__ out) {
    const int row = blockIdx.x, t = threadIdx.x;
    const float* r = m + (size_t)row * EMBED;
    __shared__ float red[256];
    float s = 0.0f;
    for (int e = t; e < EMBED; e += 256) s += r[e];
    red[t] = s; __syncthreads();
    for (int st = 128; st > 0; st >>= 1) { if (t < st) red[t] += red[t + st]; __syncthreads(); }
    const float mu = red[0] * (1.0f / EMBED);
    __syncthreads();
    float s2 = 0.0f;
    for (int e = t; e < EMBED; e += 256) { float d = r[e] - mu; s2 += d * d; }
    red[t] = s2; __syncthreads();
    for (int st = 128; st > 0; st >>= 1) { if (t < st) red[t] += red[t + st]; __syncthreads(); }
    const float rinv = rsqrtf(red[0] * (1.0f / EMBED) + LN_EPS);
    for (int e = t; e < EMBED; e += 256) {
        size_t i = (size_t)row * EMBED + e;
        out[i] = x[i] + (r[e] - mu) * rinv * g[e] + be[e];
    }
}

// ---------------------------------------------------------------------------
extern "C" void kernel_launch(void* const* d_in, const int* in_sizes, int n_in,
                              void* d_out, int out_size, void* d_ws, size_t ws_size,
                              hipStream_t stream) {
    (void)in_sizes; (void)n_in; (void)out_size; (void)ws_size;

    const float* x   = (const float*)d_in[0];
    // d_in[1] = Wq, d_in[2] = Wk : mathematically dead (softmax rows sum to 1)
    const float* Wv  = (const float*)d_in[3];
    const float* W1  = (const float*)d_in[4];
    const float* b1  = (const float*)d_in[5];
    const float* W2  = (const float*)d_in[6];
    const float* b2  = (const float*)d_in[7];
    const float* g1  = (const float*)d_in[8];
    const float* be1 = (const float*)d_in[9];
    const float* g2  = (const float*)d_in[10];
    const float* be2 = (const float*)d_in[11];
    float* out = (float*)d_out;

    // -------- workspace layout (bytes) --------
    char* w = (char*)d_ws;
    _Float16* x16 = (_Float16*)w;  w += (size_t)ROWS * EMBED * 2;   // 12.6 MB
    _Float16* Wvt = (_Float16*)w;  w += (size_t)EMBED * EMBED * 2;  //  1.2 MB
    _Float16* W1t = (_Float16*)w;  w += (size_t)HID * EMBED * 2;    //  2.4 MB
    _Float16* W2t = (_Float16*)w;  w += (size_t)EMBED * HID * 2;    //  2.4 MB
    float*    V   = (float*)w;     w += (size_t)ROWS * EMBED * 4;   // 25.2 MB (reused as m)
    float*    vs  = (float*)w;     w += (size_t)BS * EMBED * 4;
    float*    lnb = (float*)w;     w += (size_t)BS * EMBED * 4;
    _Float16* a16 = (_Float16*)w;  w += (size_t)ROWS * EMBED * 2;   // 12.6 MB
    _Float16* h16 = (_Float16*)w;  w += (size_t)ROWS * HID * 2;     // 25.2 MB
    float*    m   = V;                                              // alias: V dead after vsum

    // 1) precision conversion (+ weight transposes so WMMA B-frags are contiguous)
    cvt_f16_kernel<<<(ROWS * EMBED) / 256, 256, 0, stream>>>(x, x16, ROWS * EMBED);
    cvt_tr_kernel<<<(EMBED * EMBED) / 256, 256, 0, stream>>>(Wv, Wvt, EMBED, EMBED);
    cvt_tr_kernel<<<(EMBED * HID) / 256, 256, 0, stream>>>(W1, W1t, EMBED, HID);
    cvt_tr_kernel<<<(HID * EMBED) / 256, 256, 0, stream>>>(W2, W2t, HID, EMBED);

    // 2) V = x @ Wv   (8192 x 768 x 768, f16 WMMA, f32 accum)
    gemm_wmma<0><<<dim3(EMBED / TBN, ROWS / TBM), 256, 0, stream>>>(
        x16, Wvt, nullptr, (void*)V, ROWS, EMBED, EMBED);

    // 3) head-chunk column sums (this IS the whole attention output)
    vsum_kernel<<<dim3(12, BS), 256, 0, stream>>>(V, vs);

    // 4) LN1 of the per-batch vector; 5) a16 = f16(x + broadcast LN1)
    ln_vec_kernel<<<BS, 256, 0, stream>>>(vs, g1, be1, lnb);
    addbc_f16_kernel<<<(ROWS * EMBED) / 256, 256, 0, stream>>>(x, lnb, a16);

    // 6) h = gelu(a @ W1 + b1)  (8192 x 768 x 1536) -> f16
    gemm_wmma<1><<<dim3(HID / TBN, ROWS / TBM), 256, 0, stream>>>(
        a16, W1t, b1, (void*)h16, ROWS, HID, EMBED);

    // 7) m = gelu(h @ W2 + b2)  (8192 x 1536 x 768) -> f32
    gemm_wmma<2><<<dim3(EMBED / TBN, ROWS / TBM), 256, 0, stream>>>(
        h16, W2t, b2, (void*)m, ROWS, EMBED, HID);

    // 8) out = x + LN(m)
    final_ln_kernel<<<ROWS, 256, 0, stream>>>(x, m, g2, be2, out);
}
